// Generator_59347858096281
// MI455X (gfx1250) — matrix operand order repaired and checked by reading, not compiled, not run on hardware
//
#include <hip/hip_runtime.h>
#include <hip/hip_bf16.h>
#include <math.h>
#include <stdint.h>

// Problem constants (from reference)
#define BB   8
#define SS   256
#define HH   400
#define VV   30522
#define NSL  30
#define TT   10
#define RR   (NSL * BB)   // 240 decoder rows
#define H3   (3 * HH)     // 1200
#define KP   416          // H padded to multiple of 32 for WMMA K
#define VPAD 30720        // V padded to multiple of 512 for column blocking
#define NKSTEP (KP / 32)  // 13 WMMA K-steps

typedef __attribute__((ext_vector_type(16))) _Float16 v16h;
typedef __attribute__((ext_vector_type(8)))  float    v8f;
typedef __attribute__((ext_vector_type(4)))  float    f4;
typedef __attribute__((ext_vector_type(4)))  unsigned int u32x4;
typedef __attribute__((ext_vector_type(8)))  int      i32x8;
typedef __attribute__((ext_vector_type(4)))  int      i32x4;

union Frag16 { v16h v; f4 q[2]; };

// ---------------------------------------------------------------------------
// One-time: emb f32 [V,H] -> f16 [VPAD, KP] zero padded (rows >= V and k >= H)
// ---------------------------------------------------------------------------
__global__ __launch_bounds__(256)
void embconv_kernel(const float* __restrict__ emb, _Float16* __restrict__ embf16) {
    size_t total = (size_t)VPAD * KP;
    for (size_t i = (size_t)blockIdx.x * blockDim.x + threadIdx.x; i < total;
         i += (size_t)gridDim.x * blockDim.x) {
        size_t v = i / KP;
        int    k = (int)(i - v * KP);
        float val = (v < (size_t)VV && k < HH) ? emb[v * HH + k] : 0.0f;
        embf16[i] = (_Float16)val;
    }
}

// ---------------------------------------------------------------------------
// t=0 state init: h0[r] = X_hdd[b], x0[r] = SLOT_hdd_all[b,n,:]  (r = n*B + b)
// ---------------------------------------------------------------------------
__global__ __launch_bounds__(256)
void init_kernel(const float* __restrict__ X_hdd, const float* __restrict__ SLOT,
                 float* __restrict__ h, float* __restrict__ x) {
    int r = blockIdx.x;
    int n = r / BB, b = r % BB;
    for (int k = threadIdx.x; k < HH; k += 256) {
        h[r * HH + k] = X_hdd[b * HH + k];
        x[r * HH + k] = SLOT[((size_t)b * NSL + n) * HH + k];
    }
}

// ---------------------------------------------------------------------------
// t>=1 teacher forcing input: x[r] = emb[gold_y_idx[b, n, t-1]]
// ---------------------------------------------------------------------------
__global__ __launch_bounds__(256)
void gather_kernel(const int* __restrict__ gold, const float* __restrict__ emb,
                   float* __restrict__ x, int tm1) {
    int r = blockIdx.x;
    int n = r / BB, b = r % BB;
    int idx = gold[((size_t)b * NSL + n) * TT + tm1];
    for (int k = threadIdx.x; k < HH; k += 256)
        x[r * HH + k] = emb[(size_t)idx * HH + k];
}

// ---------------------------------------------------------------------------
// GRU cell: h <- GRU(x, h); also emit hn as f16 row padded to KP.
// One workgroup per decoder row.
// ---------------------------------------------------------------------------
__global__ __launch_bounds__(256)
void gru_kernel(const float* __restrict__ x, float* __restrict__ h,
                _Float16* __restrict__ hnf16,
                const float* __restrict__ W_ih, const float* __restrict__ W_hh,
                const float* __restrict__ b_ih, const float* __restrict__ b_hh) {
    int r = blockIdx.x;
    __shared__ float sx[HH], sh[HH], sgi[H3], sgh[H3];
    const float* xr = x + (size_t)r * HH;
    const float* hr = h + (size_t)r * HH;
    for (int i = threadIdx.x; i < HH; i += 256) { sx[i] = xr[i]; sh[i] = hr[i]; }
    __syncthreads();
    for (int j = threadIdx.x; j < H3; j += 256) {
        const float* wi = W_ih + (size_t)j * HH;
        const float* wh = W_hh + (size_t)j * HH;
        float ai = b_ih[j], ah = b_hh[j];
        for (int k = 0; k < HH; ++k) { ai += sx[k] * wi[k]; ah += sh[k] * wh[k]; }
        sgi[j] = ai; sgh[j] = ah;
    }
    __syncthreads();
    float*     hout = h + (size_t)r * HH;
    _Float16*  hf   = hnf16 + (size_t)r * KP;
    for (int k = threadIdx.x; k < HH; k += 256) {
        float rg = 1.0f / (1.0f + __expf(-(sgi[k]        + sgh[k])));
        float zg = 1.0f / (1.0f + __expf(-(sgi[HH + k]   + sgh[HH + k])));
        float ng = tanhf(sgi[2 * HH + k] + rg * sgh[2 * HH + k]);
        float hn = (1.0f - zg) * ng + zg * sh[k];
        hout[k] = hn;
        hf[k]   = (_Float16)hn;
    }
    if (threadIdx.x < (KP - HH)) hf[HH + threadIdx.x] = (_Float16)0.0f;
}

// ---------------------------------------------------------------------------
// Attention + pointer switch: scores over S, masked softmax -> prob, ctx,
// sw = sigmoid(W_ratio . [hn, ctx, x] + b_ratio). One workgroup per row.
// blockDim.x == SS == 256.
// ---------------------------------------------------------------------------
__global__ __launch_bounds__(256)
void attn_kernel(const float* __restrict__ h, const float* __restrict__ x,
                 const float* __restrict__ enc, const int* __restrict__ input_len,
                 const float* __restrict__ W_ratio, const float* __restrict__ b_ratio,
                 float* __restrict__ prob, float* __restrict__ sw) {
    int r = blockIdx.x;
    int b = r % BB;
    __shared__ float shn[HH];
    __shared__ float sprob[SS];
    __shared__ float red[256];
    const float* hr = h + (size_t)r * HH;
    for (int i = threadIdx.x; i < HH; i += 256) shn[i] = hr[i];
    __syncthreads();

    int s = threadIdx.x;
    const float* es = enc + ((size_t)b * SS + s) * HH;
    float sc = 0.0f;
    for (int k = 0; k < HH; ++k) sc += shn[k] * es[k];
    int len = input_len[b]; if (len < 1) len = 1;
    if (s >= len) sc = -1e9f;

    red[s] = sc; __syncthreads();
    for (int off = 128; off > 0; off >>= 1) {
        if (s < off) red[s] = fmaxf(red[s], red[s + off]);
        __syncthreads();
    }
    float mx = red[0]; __syncthreads();
    float e = __expf(sc - mx);
    red[s] = e; __syncthreads();
    for (int off = 128; off > 0; off >>= 1) {
        if (s < off) red[s] = red[s] + red[s + off];
        __syncthreads();
    }
    float p = e / red[0];
    __syncthreads();
    sprob[s] = p;
    prob[(size_t)r * SS + s] = p;
    __syncthreads();

    // ctx + pointer-switch partial
    float part = 0.0f;
    for (int hh = threadIdx.x; hh < HH; hh += 256) {
        float c = 0.0f;
        const float* col = enc + (size_t)b * SS * HH + hh;
        for (int ss2 = 0; ss2 < SS; ++ss2) c += sprob[ss2] * col[(size_t)ss2 * HH];
        part += W_ratio[HH + hh] * c;          // ctx slice
        part += W_ratio[hh] * shn[hh];         // hn slice
        part += W_ratio[2 * HH + hh] * x[(size_t)r * HH + hh];  // x slice
    }
    red[threadIdx.x] = part; __syncthreads();
    for (int off = 128; off > 0; off >>= 1) {
        if (threadIdx.x < off) red[threadIdx.x] += red[threadIdx.x + off];
        __syncthreads();
    }
    if (threadIdx.x == 0)
        sw[r] = 1.0f / (1.0f + __expf(-(red[0] + b_ratio[0])));
}

// ---------------------------------------------------------------------------
// Vocab GEMM with WMMA: logits[240, VPAD] = hnf16[240, KP] @ embf16[VPAD, KP]^T
// grid = (VPAD/512, 15); block = 256 (8 waves). Each wave: one 16x64 tile.
//
// The 16xKP A tile is shared by all 8 waves: wave 0 issues one Tensor Data
// Mover load (tensor_load_to_lds, 2-D descriptor per CDNA5 ISA),
// waits on TENSORcnt, and a workgroup barrier publishes it. A fragments then
// come from LDS (ds_load, DScnt) while B fragments stream from L2 (loadcnt),
// decoupling the two dependency chains. B is ping-pong double-buffered.
// Fragment packing: A and B both use contiguous k = 16*grp + i per lane.
// ---------------------------------------------------------------------------
__global__ __launch_bounds__(256, 1)
void vocab_gemm_kernel(const _Float16* __restrict__ hnf16,
                       const _Float16* __restrict__ embf16,
                       float* __restrict__ logits) {
    __shared__ _Float16 smemA[16 * KP];

    const int lane = threadIdx.x & 31;
    const int wave = threadIdx.x >> 5;
    const int grp  = lane >> 4;       // lane group 0/1
    const int ml   = lane & 15;

    const int rowTile = blockIdx.y;                        // 0..14
    const int colBase = blockIdx.x * 512 + wave * 64;      // 64 cols per wave

    // --- TDM: pull the 16 x KP f16 A tile into LDS (wave 0 only) ---
    if (threadIdx.x < 32) {
        const uint64_t ga = (uint64_t)(uintptr_t)(hnf16 + (size_t)rowTile * 16 * KP);
        // Flat LDS pointer: addr[31:0] is the LDS offset (ISA §10.2 aperture).
        const uint32_t ldsAddr = (uint32_t)(uintptr_t)(&smemA[0]);

        u32x4 g0;
        g0[0] = 1u;                                            // count=1 (valid), load, user
        g0[1] = ldsAddr;                                       // lds_addr (bytes)
        g0[2] = (uint32_t)ga;                                  // global_addr[31:0]
        g0[3] = (uint32_t)((ga >> 32) & 0x01FFFFFFu)           // global_addr[56:32]
              | (2u << 30);                                    // type = 2 ("image")

        i32x8 g1;
        g1[0] = (int)(1u << 16);                               // data_size=1 -> 2 bytes
        g1[1] = (int)((uint32_t)(KP & 0xFFFF) << 16);          // tensor_dim0[15:0]
        g1[2] = (int)(((uint32_t)KP >> 16) & 0xFFFFu)          // tensor_dim0[31:16]
              | (int)((uint32_t)(RR & 0xFFFF) << 16);          // tensor_dim1[15:0]
        g1[3] = (int)(((uint32_t)RR >> 16) & 0xFFFFu)          // tensor_dim1[31:16]
              | (int)((uint32_t)KP << 16);                     // tile_dim0 = KP
        g1[4] = 16;                                            // tile_dim1=16, tile_dim2=0
        g1[5] = KP;                                            // tensor_dim0_stride[31:0]
        g1[6] = 0;                                             // stride high bits
        g1[7] = 0;

        i32x4 gz;
        gz[0] = 0; gz[1] = 0; gz[2] = 0; gz[3] = 0;            // 2-D tensor: groups 2/3 unused
        i32x8 gz8;
        gz8[0] = 0; gz8[1] = 0; gz8[2] = 0; gz8[3] = 0;
        gz8[4] = 0; gz8[5] = 0; gz8[6] = 0; gz8[7] = 0;

        __builtin_amdgcn_tensor_load_to_lds(g0, g1, gz, gz, gz8, 0);
        __builtin_amdgcn_s_wait_tensorcnt(0);
    }
    __syncthreads();

    const _Float16* Arow = &smemA[ml * KP];                    // LDS A row for this lane
    const _Float16* Bcol[4];
#pragma unroll
    for (int j = 0; j < 4; ++j)
        Bcol[j] = embf16 + (size_t)(colBase + j * 16 + ml) * KP + 16 * grp;

    v8f acc[4];
#pragma unroll
    for (int j = 0; j < 4; ++j)
#pragma unroll
        for (int i = 0; i < 8; ++i) acc[j][i] = 0.0f;

    // Ping-pong B fragment buffers (A comes from LDS each step)
    Frag16 bfr[2][4];
#pragma unroll
    for (int j = 0; j < 4; ++j) {
        bfr[0][j].q[0] = *(const f4*)(Bcol[j]);
        bfr[0][j].q[1] = *(const f4*)(Bcol[j] + 8);
    }

#pragma unroll
    for (int s = 0; s < NKSTEP; ++s) {
        const int cur = s & 1;
        const int nxt = cur ^ 1;
        if (s + 1 < NKSTEP) {
            const int kb2 = (s + 1) * 32;
#pragma unroll
            for (int j = 0; j < 4; ++j) {
                bfr[nxt][j].q[0] = *(const f4*)(Bcol[j] + kb2);
                bfr[nxt][j].q[1] = *(const f4*)(Bcol[j] + kb2 + 8);
            }
        }
        // A fragment from LDS: halves 0..15 -> K = kb + 16*grp + i
        Frag16 a;
        const _Float16* pa = Arow + s * 32 + 16 * grp;
        a.q[0] = *(const f4*)(pa);
        a.q[1] = *(const f4*)(pa + 8);
#pragma unroll
        for (int j = 0; j < 4; ++j) {
            acc[j] = __builtin_amdgcn_wmma_f32_16x16x32_f16(
                         false, a.v, false, bfr[cur][j].v,
                         (short)0, acc[j], false, false);
        }
    }

    // D layout: VGPR row rr -> M = rr + 8*grp, N = ml
#pragma unroll
    for (int j = 0; j < 4; ++j) {
        const int ncol = colBase + j * 16 + ml;
#pragma unroll
        for (int rr = 0; rr < 8; ++rr) {
            const int m = rr + 8 * grp;
            logits[(size_t)(rowTile * 16 + m) * VPAD + ncol] = acc[j][rr];
        }
    }
}

// ---------------------------------------------------------------------------
// Per-row softmax statistics over V (max, sum of exp). One workgroup per row.
// ---------------------------------------------------------------------------
__global__ __launch_bounds__(256)
void rowstat_kernel(const float* __restrict__ logits,
                    float* __restrict__ rowmax, float* __restrict__ rowsum) {
    int r = blockIdx.x;
    const float* L = logits + (size_t)r * VPAD;
    __shared__ float red[256];
    float mx = -1e30f;
    for (int v = threadIdx.x; v < VV; v += 256) mx = fmaxf(mx, L[v]);
    red[threadIdx.x] = mx; __syncthreads();
    for (int off = 128; off > 0; off >>= 1) {
        if (threadIdx.x < off) red[threadIdx.x] = fmaxf(red[threadIdx.x], red[threadIdx.x + off]);
        __syncthreads();
    }
    mx = red[0]; __syncthreads();
    float s = 0.0f;
    for (int v = threadIdx.x; v < VV; v += 256) s += __expf(L[v] - mx);
    red[threadIdx.x] = s; __syncthreads();
    for (int off = 128; off > 0; off >>= 1) {
        if (threadIdx.x < off) red[threadIdx.x] += red[threadIdx.x + off];
        __syncthreads();
    }
    if (threadIdx.x == 0) { rowmax[r] = mx; rowsum[r] = red[0]; }
}

// ---------------------------------------------------------------------------
// final = sw * p_vocab  (full coverage of out slice for step t; scatter adds
// later). Non-temporal load/store: logits row is dead after this read and the
// 293 MB output is write-once -> keep L2 free for embf16 + next logits tile.
// grid = (ceil(V/256), 240)
// ---------------------------------------------------------------------------
__global__ __launch_bounds__(256)
void final_kernel(const float* __restrict__ logits, const float* __restrict__ rowmax,
                  const float* __restrict__ rowsum, const float* __restrict__ sw,
                  float* __restrict__ out, int t) {
    int v = blockIdx.x * 256 + threadIdx.x;
    int r = blockIdx.y;
    if (v >= VV) return;
    float scale = sw[r] / rowsum[r];
    float lg = __builtin_nontemporal_load(&logits[(size_t)r * VPAD + v]);
    float p = __expf(lg - rowmax[r]) * scale;
    __builtin_nontemporal_store(p, &out[((size_t)r * TT + t) * VV + v]);
}

// ---------------------------------------------------------------------------
// Pointer scatter: out[r, t, story[b,s]] += (1-sw[r]) * prob[r,s]
// ---------------------------------------------------------------------------
__global__ __launch_bounds__(256)
void scatter_kernel(const float* __restrict__ prob, const float* __restrict__ sw,
                    const int* __restrict__ input_seq, float* __restrict__ out, int t) {
    int r = blockIdx.x;
    int b = r % BB;
    int s = threadIdx.x;          // SS == 256 == blockDim
    int v = input_seq[(size_t)b * SS + s];
    float val = (1.0f - sw[r]) * prob[(size_t)r * SS + s];
    atomicAdd(&out[((size_t)r * TT + t) * VV + v], val);
}

// ---------------------------------------------------------------------------
extern "C" void kernel_launch(void* const* d_in, const int* in_sizes, int n_in,
                              void* d_out, int out_size, void* d_ws, size_t ws_size,
                              hipStream_t stream) {
    (void)in_sizes; (void)n_in; (void)out_size; (void)ws_size;

    const int*   input_seq = (const int*)  d_in[0];
    const int*   input_len = (const int*)  d_in[1];
    const float* enc       = (const float*)d_in[2];
    const float* X_hdd     = (const float*)d_in[3];
    const float* SLOT      = (const float*)d_in[4];
    const int*   gold      = (const int*)  d_in[5];
    const float* emb       = (const float*)d_in[6];
    const float* W_ih      = (const float*)d_in[7];
    const float* W_hh      = (const float*)d_in[8];
    const float* b_ih      = (const float*)d_in[9];
    const float* b_hh      = (const float*)d_in[10];
    const float* W_ratio   = (const float*)d_in[11];
    const float* b_ratio   = (const float*)d_in[12];
    float*       out       = (float*)d_out;

    // workspace carve-up (256B aligned slices)
    char* ws = (char*)d_ws;
    size_t off = 0;
    auto take = [&](size_t bytes) -> void* {
        void* p = (void*)(ws + off);
        off += (bytes + 255) & ~(size_t)255;
        return p;
    };
    _Float16* embf16 = (_Float16*)take((size_t)VPAD * KP * sizeof(_Float16)); // 25.6 MB
    float*    logits = (float*)   take((size_t)RR * VPAD * sizeof(float));    // 29.5 MB
    _Float16* hnf16  = (_Float16*)take((size_t)RR * KP * sizeof(_Float16));
    float*    hbuf   = (float*)   take((size_t)RR * HH * sizeof(float));
    float*    xbuf   = (float*)   take((size_t)RR * HH * sizeof(float));
    float*    prob   = (float*)   take((size_t)RR * SS * sizeof(float));
    float*    sw     = (float*)   take((size_t)RR * sizeof(float));
    float*    rowmax = (float*)   take((size_t)RR * sizeof(float));
    float*    rowsum = (float*)   take((size_t)RR * sizeof(float));

    embconv_kernel<<<4096, 256, 0, stream>>>(emb, embf16);
    init_kernel<<<RR, 256, 0, stream>>>(X_hdd, SLOT, hbuf, xbuf);

    for (int t = 0; t < TT; ++t) {
        if (t > 0)
            gather_kernel<<<RR, 256, 0, stream>>>(gold, emb, xbuf, t - 1);
        gru_kernel<<<RR, 256, 0, stream>>>(xbuf, hbuf, hnf16, W_ih, W_hh, b_ih, b_hh);
        attn_kernel<<<RR, 256, 0, stream>>>(hbuf, xbuf, enc, input_len,
                                            W_ratio, b_ratio, prob, sw);
        dim3 gg(VPAD / 512, 15);
        vocab_gemm_kernel<<<gg, 256, 0, stream>>>(hnf16, embf16, logits);
        rowstat_kernel<<<RR, 256, 0, stream>>>(logits, rowmax, rowsum);
        dim3 gf((VV + 255) / 256, RR);
        final_kernel<<<gf, 256, 0, stream>>>(logits, rowmax, rowsum, sw, out, t);
        scatter_kernel<<<RR, 256, 0, stream>>>(prob, sw, input_seq, out, t);
    }
}
